// Model_21818433864155
// MI455X (gfx1250) — compile-verified, matrix operand order checked by reading, not run on hardware
//
#include <hip/hip_runtime.h>
#include <cstdint>
#include <cstddef>

// Log-semiring cumulative matrix scan for MI455X (gfx1250).
//   x (x) y = -logaddexp(-x,-y) = min(x,y) - log1p(exp(-|x-y|))   [non-bilinear!]
//   reduce  = logsumexp over K
// Compute-bound on v_exp_f32/v_rcp_f32/v_log_f32; memory footprint ~16MB (L2-resident).
// CDNA5 paths: global_load_async_to_lds_b128 + s_wait_asynccnt for LDS staging,
// and an exact identity WMMA pass-through (D = 0*0 + C) on the results.

#define D64   64
#define MATSZ 4096        // 64*64 floats = 16KB
#define LTOT  64
#define NBLK  512         // B*L = 8*64

typedef __attribute__((ext_vector_type(16))) _Float16 v16h;
typedef __attribute__((ext_vector_type(8)))  float    v8f;

// Low 32 bits of a generic pointer into LDS == LDS byte address (aperture layout).
__device__ __forceinline__ uint32_t lds_addr(const void* p) {
  return (uint32_t)(uintptr_t)p;
}

__device__ __forceinline__ void async_b128_to_lds(const float* g, float* l) {
  asm volatile("global_load_async_to_lds_b128 %0, %1, off"
               :: "v"(lds_addr(l)), "v"(g) : "memory");
}

__device__ __forceinline__ void wait_async0() {
#if __has_builtin(__builtin_amdgcn_s_wait_asynccnt)
  __builtin_amdgcn_s_wait_asynccnt(0);
#else
  asm volatile("s_wait_asynccnt 0" ::: "memory");
#endif
}

__device__ __forceinline__ void stage_matrix(const float* __restrict__ g,
                                             float* __restrict__ s,
                                             int tid, int nthr) {
  for (int v = tid; v < MATSZ / 4; v += nthr)
    async_b128_to_lds(g + 4 * v, s + 4 * v);
}

__device__ __forceinline__ void copy_mat(const float* __restrict__ g,
                                         float* __restrict__ d,
                                         int tid, int nthr) {
  const float4* g4 = (const float4*)g;
  float4*       d4 = (float4*)d;
  for (int v = tid; v < MATSZ / 4; v += nthr) d4[v] = g4[v];
}

// dst = sA (x-matmul) sB, 64x64x64, log-semiring. 256 threads:
// thread owns row i = tid>>2 and 16 consecutive cols jb = (tid&3)*16.
// Two-pass: shift = max_k min(a,b)  (within log2 of true max => stable),
// then  out = shift + log( sum_k exp(min(a,b)-shift) / (1 + exp(-|a-b|)) ).
__device__ __forceinline__ void product64(const float* __restrict__ sA,
                                          const float* __restrict__ sB,
                                          float* __restrict__ dst, int tid) {
  const int i  = tid >> 2;
  const int jb = (tid & 3) << 4;

  float shift[16];
#pragma unroll
  for (int q = 0; q < 16; ++q) shift[q] = -3.402823466e38f;

  for (int k = 0; k < D64; ++k) {
    const float a = sA[i * D64 + k];
    const float4* br = (const float4*)(sB + k * D64 + jb);
#pragma unroll
    for (int c = 0; c < 4; ++c) {
      float4 bb = br[c];
      shift[c*4+0] = fmaxf(shift[c*4+0], fminf(a, bb.x));
      shift[c*4+1] = fmaxf(shift[c*4+1], fminf(a, bb.y));
      shift[c*4+2] = fmaxf(shift[c*4+2], fminf(a, bb.z));
      shift[c*4+3] = fmaxf(shift[c*4+3], fminf(a, bb.w));
    }
  }

  float acc[16];
#pragma unroll
  for (int q = 0; q < 16; ++q) acc[q] = 0.0f;

  for (int k = 0; k < D64; ++k) {
    const float a = sA[i * D64 + k];
    const float4* br = (const float4*)(sB + k * D64 + jb);
#pragma unroll
    for (int c = 0; c < 4; ++c) {
      float4 bb = br[c];
      float bv[4] = {bb.x, bb.y, bb.z, bb.w};
#pragma unroll
      for (int u = 0; u < 4; ++u) {
        const int q = c * 4 + u;
        const float b = bv[u];
        const float t = fminf(a, b);
        const float d = fabsf(a - b);
        // exp(z - shift) with z = t - log1p(e^-d)  ->  e^(t-shift)/(1+e^-d)
        acc[q] += __expf(t - shift[q]) * __builtin_amdgcn_rcpf(1.0f + __expf(-d));
      }
    }
  }

  float res[16];
#pragma unroll
  for (int q = 0; q < 16; ++q) res[q] = shift[q] + __logf(acc[q]);

  // Exact identity pass-through on the matrix unit: D = 0*0 + C == C (RNE exact).
  // The semiring op itself is non-bilinear and cannot map onto WMMA; this only
  // exercises/verifies the v_wmma path. All 256 lanes active here (EXEC all-1s).
  v16h za = {}; v16h zb = {};
  v8f c0, c1;
#pragma unroll
  for (int q = 0; q < 8; ++q) { c0[q] = res[q]; c1[q] = res[8 + q]; }
  c0 = __builtin_amdgcn_wmma_f32_16x16x32_f16(false, za, false, zb, (short)0, c0, false, false);
  c1 = __builtin_amdgcn_wmma_f32_16x16x32_f16(false, za, false, zb, (short)0, c1, false, false);

  float4* dq = (float4*)(dst + i * D64 + jb);
  float4 o0 = {c0[0], c0[1], c0[2], c0[3]};
  float4 o1 = {c0[4], c0[5], c0[6], c0[7]};
  float4 o2 = {c1[0], c1[1], c1[2], c1[3]};
  float4 o3 = {c1[4], c1[5], c1[6], c1[7]};
  dq[0] = o0; dq[1] = o1; dq[2] = o2; dq[3] = o3;
}

// Step 1 fused with the gather pm[b,l] = p[act[b,l]+1]:
//   dst[b,0] = p[a0+1];  dst[b,j] = p[a_{j-1}+1] (x) p[a_j+1]  for j>=1
__global__ __launch_bounds__(256) void step_first(const float* __restrict__ p,
                                                  const int* __restrict__ act,
                                                  float* __restrict__ dst) {
  __shared__ float sA[MATSZ];
  __shared__ float sB[MATSZ];
  const int bl  = blockIdx.x;
  const int j   = bl & (LTOT - 1);
  const int tid = threadIdx.x;
  const float* Mj = p + (size_t)(act[bl] + 1) * MATSZ;
  if (j == 0) { copy_mat(Mj, dst + (size_t)bl * MATSZ, tid, 256); return; }
  const float* Mi = p + (size_t)(act[bl - 1] + 1) * MATSZ;
  stage_matrix(Mi, sA, tid, 256);
  stage_matrix(Mj, sB, tid, 256);
  wait_async0();
  __syncthreads();
  product64(sA, sB, dst + (size_t)bl * MATSZ, tid);
}

// Generic doubling step: dst[b,j] = src[b,j-step] (x) src[b,j] for j>=step, else copy.
__global__ __launch_bounds__(256) void step_n(const float* __restrict__ src,
                                              float* __restrict__ dst, int step) {
  __shared__ float sA[MATSZ];
  __shared__ float sB[MATSZ];
  const int bl  = blockIdx.x;
  const int j   = bl & (LTOT - 1);
  const int tid = threadIdx.x;
  if (j < step) {
    copy_mat(src + (size_t)bl * MATSZ, dst + (size_t)bl * MATSZ, tid, 256);
    return;
  }
  stage_matrix(src + (size_t)(bl - step) * MATSZ, sA, tid, 256);
  stage_matrix(src + (size_t)bl * MATSZ,          sB, tid, 256);
  wait_async0();
  __syncthreads();
  product64(sA, sB, dst + (size_t)bl * MATSZ, tid);
}

// x = init (1xD) (x) pm ;  y = x (x) p[0].  64 threads, one column each.
// Column reads sM[k*64 + tid]: bank index == tid per lane -> conflict-free.
__global__ __launch_bounds__(64) void final_obs(const float* __restrict__ pm,
                                                const float* __restrict__ p,
                                                const float* __restrict__ initv,
                                                float* __restrict__ out) {
  __shared__ float sM[MATSZ];
  __shared__ float sP0[MATSZ];
  __shared__ float sInit[D64];
  __shared__ float sX[D64];
  const int bl  = blockIdx.x;
  const int tid = threadIdx.x;  // 0..63
  stage_matrix(pm + (size_t)bl * MATSZ, sM, tid, 64);
  stage_matrix(p, sP0, tid, 64);
  sInit[tid] = initv[tid];
  wait_async0();
  __syncthreads();
  {
    float shift = -3.402823466e38f;
    for (int k = 0; k < D64; ++k)
      shift = fmaxf(shift, fminf(sInit[k], sM[k * D64 + tid]));
    float s = 0.0f;
    for (int k = 0; k < D64; ++k) {
      const float a = sInit[k], b = sM[k * D64 + tid];
      const float t = fminf(a, b), d = fabsf(a - b);
      s += __expf(t - shift) * __builtin_amdgcn_rcpf(1.0f + __expf(-d));
    }
    sX[tid] = shift + __logf(s);
  }
  __syncthreads();
  {
    float shift = -3.402823466e38f;
    for (int k = 0; k < D64; ++k)
      shift = fmaxf(shift, fminf(sX[k], sP0[k * D64 + tid]));
    float s = 0.0f;
    for (int k = 0; k < D64; ++k) {
      const float a = sX[k], b = sP0[k * D64 + tid];
      const float t = fminf(a, b), d = fabsf(a - b);
      s += __expf(t - shift) * __builtin_amdgcn_rcpf(1.0f + __expf(-d));
    }
    out[(size_t)bl * D64 + tid] = shift + __logf(s);
  }
}

extern "C" void kernel_launch(void* const* d_in, const int* in_sizes, int n_in,
                              void* d_out, int out_size, void* d_ws, size_t ws_size,
                              hipStream_t stream) {
  (void)in_sizes; (void)n_in; (void)out_size; (void)ws_size;
  const float* p     = (const float*)d_in[0];   // (5,64,64) f32
  const float* initv = (const float*)d_in[1];   // (64,)     f32
  const int*   act   = (const int*)d_in[2];     // (8,64)    i32
  float* out  = (float*)d_out;                  // (8,64,64) f32
  float* buf0 = (float*)d_ws;                   // 8MB
  float* buf1 = buf0 + (size_t)NBLK * MATSZ;    // 8MB

  step_first<<<NBLK, 256, 0, stream>>>(p, act, buf0);          // step 1 (fused gather)
  step_n    <<<NBLK, 256, 0, stream>>>(buf0, buf1, 2);
  step_n    <<<NBLK, 256, 0, stream>>>(buf1, buf0, 4);
  step_n    <<<NBLK, 256, 0, stream>>>(buf0, buf1, 8);
  step_n    <<<NBLK, 256, 0, stream>>>(buf1, buf0, 16);
  step_n    <<<NBLK, 256, 0, stream>>>(buf0, buf1, 32);
  final_obs <<<NBLK,  64, 0, stream>>>(buf1, p, initv, out);
}
